// MultiMolNet_42941083025858
// MI455X (gfx1250) — matrix operand order repaired and checked by reading, not compiled, not run on hardware
//
#include <hip/hip_runtime.h>
#include <hip/hip_bf16.h>

// Problem constants (from reference)
#define NB   4096      // molecules
#define NA   64        // atoms per molecule
#define NF   124       // features
#define NFP  128       // padded K for layer 1
#define NT   8         // types
#define NH1  64
#define NH2  16
#define NATOM (NB*NA)  // 262144
#define NW   8         // waves per block in main kernel
#define MAXTILES (NATOM/16 + NT)   // 16392 upper bound on Σ ceil(cnt_t/16)

typedef float  v8f   __attribute__((ext_vector_type(8)));
typedef __bf16 v16bf __attribute__((ext_vector_type(16)));
typedef __bf16 v8bf  __attribute__((ext_vector_type(8)));

// ---------------- workspace layout ----------------
// [0)      int counts[8]
// [32)     int bases[8]
// [64)     int cursors[8]
// [96)     float C[8]
// [128)    float S
// [256)    bf16 W1T[8][64][128]   (N-major, K padded)   131072 B
// [131328) bf16 W2T[8][16][64]    (N-major)             16384 B
// [147712) int  atom_list[NATOM]                        1048576 B
// [1196288) float per_atom[NATOM]                       1048576 B

// ---------- k_init: zero counters, per-type zero-input constants ----------
__global__ void k_init(const float* __restrict__ b1, const float* __restrict__ W2,
                       const float* __restrict__ b2, const float* __restrict__ W3,
                       const float* __restrict__ b3,
                       int* counts, float* C, float* S) {
  __shared__ float sc[NT];
  int tid = threadIdx.x;
  if (tid < NT) counts[tid] = 0;
  if (tid < NT) {
    int t = tid;
    float h1[NH1];
    #pragma unroll
    for (int h = 0; h < NH1; h++) { float v = b1[t*NH1 + h]; h1[h] = v > 0.f ? v : 0.f; }
    float o = b3[t];
    for (int g = 0; g < NH2; g++) {
      float a = b2[t*NH2 + g];
      for (int h = 0; h < NH1; h++) a += h1[h] * W2[(t*NH1 + h)*NH2 + g];
      a = a > 0.f ? a : 0.f;
      o += a * W3[t*NH2 + g];
    }
    C[t] = o;
    sc[t] = o;
  }
  __syncthreads();
  if (tid == 0) {
    float s = 0.f;
    #pragma unroll
    for (int i = 0; i < NT; i++) s += sc[i];
    *S = s;
  }
}

// ---------- weight transpose/convert to WMMA B layout ----------
__global__ void k_w1t(const float* __restrict__ W1, __bf16* __restrict__ W1T) {
  int idx = blockIdx.x * blockDim.x + threadIdx.x;      // [0, 8*64*128)
  if (idx >= NT*NH1*NFP) return;
  int t = idx >> 13;          // /8192
  int rem = idx & 8191;
  int h = rem >> 7;           // N (output column)
  int k = rem & 127;          // K (feature)
  float v = (k < NF) ? W1[(t*NF + k)*NH1 + h] : 0.f;
  W1T[idx] = (__bf16)v;
}

__global__ void k_w2t(const float* __restrict__ W2, __bf16* __restrict__ W2T) {
  int idx = blockIdx.x * blockDim.x + threadIdx.x;      // [0, 8*16*64)
  if (idx >= NT*NH2*NH1) return;
  int t = idx >> 10;
  int rem = idx & 1023;
  int g = rem >> 6;           // N
  int h = rem & 63;           // K
  W2T[idx] = (__bf16)W2[(t*NH1 + h)*NH2 + g];
}

// ---------- bucketing ----------
__global__ void k_hist(const float* __restrict__ x, int* counts) {
  int id = blockIdx.x * blockDim.x + threadIdx.x;
  if (id >= NATOM) return;
  int t = (int)x[(long)id * (NF + 1)];
  atomicAdd(&counts[t], 1);
}

__global__ void k_bases(const int* counts, int* bases, int* cursors) {
  if (threadIdx.x == 0 && blockIdx.x == 0) {
    int acc = 0;
    for (int t = 0; t < NT; t++) { bases[t] = acc; cursors[t] = acc; acc += counts[t]; }
  }
}

__global__ void k_scatter(const float* __restrict__ x, int* cursors, int* __restrict__ list) {
  int id = blockIdx.x * blockDim.x + threadIdx.x;
  if (id >= NATOM) return;
  int t = (int)x[(long)id * (NF + 1)];
  int slot = atomicAdd(&cursors[t], 1);
  list[slot] = id;
}

// ---------- main MLP kernel: one wave per 16-atom tile ----------
__global__ __launch_bounds__(256) void k_mlp(
    const float* __restrict__ x, const float* __restrict__ mask,
    const float* __restrict__ b1, const float* __restrict__ b2,
    const float* __restrict__ W3, const float* __restrict__ b3,
    const int* __restrict__ counts, const int* __restrict__ bases,
    const float* __restrict__ Cs, const float* __restrict__ Sp,
    const __bf16* __restrict__ W1T, const __bf16* __restrict__ W2T,
    const int* __restrict__ list, float* __restrict__ per_atom) {

  __shared__ __bf16 sA[NW][16][NFP];   // 4 KB/wave : features (bf16, K-padded)
  __shared__ __bf16 sH[NW][16][NH1];   // 2 KB/wave : layer-1 activations
  __shared__ int    sAtom[NW][16];
  __shared__ float  sMask[NW][16];
  __shared__ int    sValid[NW][16];

  const int w    = threadIdx.x >> 5;
  const int lane = threadIdx.x & 31;
  const int tile = blockIdx.x * NW + w;

  // Locate bucket (type) and local tile index.
  int t = -1, local = 0, cnt_t = 0, base_t = 0, acc = 0;
  #pragma unroll
  for (int i = 0; i < NT; i++) {
    int c  = counts[i];
    int nt = (c + 15) >> 4;
    if (t < 0 && tile < acc + nt) { t = i; local = tile - acc; cnt_t = c; base_t = bases[i]; }
    acc += nt;
  }
  const bool live = (t >= 0);
  if (!live) { t = 0; local = 0; cnt_t = counts[0]; base_t = bases[0]; }

  // Stage features: 2 lanes per row, 62 floats each; convert f32->bf16 into LDS.
  const int myrow = lane >> 1, half = lane & 1;
  int ib = local * 16 + myrow;
  const bool valid = live && (ib < cnt_t);
  int li = base_t + (valid ? ib : 0);
  if (li > NATOM - 1) li = NATOM - 1;
  const int atom = list[li];
  const float* xa = x + (long)atom * (NF + 1) + 1;
  #pragma unroll
  for (int j = 0; j < 62; j++) {
    int k = half * 62 + j;
    sA[w][myrow][k] = (__bf16)xa[k];
  }
  if (!half) {
    sA[w][myrow][124] = (__bf16)0.f;
    sA[w][myrow][125] = (__bf16)0.f;
    sA[w][myrow][126] = (__bf16)0.f;
    sA[w][myrow][127] = (__bf16)0.f;
    sAtom[w][myrow]  = atom;
    sMask[w][myrow]  = mask[atom];
    sValid[w][myrow] = valid ? 1 : 0;
  }
  __syncthreads();   // all waves reach here (no early returns)

  // A-fragment lane mapping (16-bit A 16x32): lane = M; lanes<16 hold K {0..7,16..23},
  // lanes>=16 hold K {8..15,24..31}.
  const int M  = lane & 15;
  const int ko = (lane < 16) ? 0 : 8;
  // B-fragment lane mapping: lane = N; same K order as A (lanes<16 K {0..7,16..23}, lanes>=16 K {8..15,24..31}).
  const int n  = lane & 15;
  const int kb = (lane < 16) ? 0 : 16;
  const int mB = (lane < 16) ? 0 : 8;

  // Load all four K-chunks of A once (4 x 8 VGPRs bf16).
  v16bf a1[4];
  #pragma unroll
  for (int ks = 0; ks < 4; ks++) {
    const __bf16* pr = &sA[w][M][ks * 32];
    v8bf lo = *(const v8bf*)(pr + ko);
    v8bf hi = *(const v8bf*)(pr + 16 + ko);
    #pragma unroll
    for (int i = 0; i < 8; i++) { a1[ks][i] = lo[i]; a1[ks][i + 8] = hi[i]; }
  }

  // ---- Layer 1: 16x128 @ 128x64 -> 16x64, four 16-col groups x four K-steps.
  const __bf16* w1t = W1T + (long)t * NH1 * NFP;
  #pragma unroll
  for (int ng = 0; ng < 4; ng++) {
    v8f c = {0.f,0.f,0.f,0.f,0.f,0.f,0.f,0.f};
    #pragma unroll
    for (int ks = 0; ks < 4; ks++) {
      v8bf blo = *(const v8bf*)(w1t + (ng * 16 + n) * NFP + ks * 32 + ko); v8bf bhi = *(const v8bf*)(w1t + (ng * 16 + n) * NFP + ks * 32 + 16 + ko); v16bf bm; for (int i = 0; i < 8; i++) { bm[i] = blo[i]; bm[i + 8] = bhi[i]; }
      c = __builtin_amdgcn_wmma_f32_16x16x32_bf16(false, a1[ks], false, bm,
                                                  (short)0, c, false, false);
    }
    const float bias = b1[t * NH1 + ng * 16 + n];
    #pragma unroll
    for (int r = 0; r < 8; r++) {
      float hv = c[r] + bias;
      hv = hv > 0.f ? hv : 0.f;
      sH[w][mB + r][ng * 16 + n] = (__bf16)hv;   // C/D layout: lane=N, VGPR r -> M=mB+r
    }
  }
  __syncthreads();

  // ---- Layer 2: 16x64 @ 64x16 -> 16x16, two K-steps.
  const __bf16* w2t = W2T + (long)t * NH2 * NH1;
  v8f c2 = {0.f,0.f,0.f,0.f,0.f,0.f,0.f,0.f};
  #pragma unroll
  for (int ks = 0; ks < 2; ks++) {
    const __bf16* pr = &sH[w][M][ks * 32];
    v8bf lo = *(const v8bf*)(pr + ko);
    v8bf hi = *(const v8bf*)(pr + 16 + ko);
    v16bf a2;
    #pragma unroll
    for (int i = 0; i < 8; i++) { a2[i] = lo[i]; a2[i + 8] = hi[i]; }
    v8bf blo = *(const v8bf*)(w2t + n * NH1 + ks * 32 + ko); v8bf bhi = *(const v8bf*)(w2t + n * NH1 + ks * 32 + 16 + ko); v16bf bm; for (int i = 0; i < 8; i++) { bm[i] = blo[i]; bm[i + 8] = bhi[i]; }
    c2 = __builtin_amdgcn_wmma_f32_16x16x32_bf16(false, a2, false, bm,
                                                 (short)0, c2, false, false);
  }

  // ---- Layer 3: per-lane multiply by W3[t][n], reduce across 16 lanes per row.
  const float bias2 = b2[t * NH2 + n];
  const float w3v   = W3[t * NH2 + n];
  const float corr  = *Sp - Cs[t];      // Σ_t C_t − C_type
  const float b3v   = b3[t];
  float vs[8];
  #pragma unroll
  for (int r = 0; r < 8; r++) {
    float h2 = c2[r] + bias2;
    h2 = h2 > 0.f ? h2 : 0.f;
    float q = h2 * w3v;
    q += __shfl_xor(q, 1, 16);
    q += __shfl_xor(q, 2, 16);
    q += __shfl_xor(q, 4, 16);
    q += __shfl_xor(q, 8, 16);
    vs[r] = q;                           // row M = mB + r sum, replicated in 16-lane group
  }

  if (live && (lane == 0 || lane == 16)) {
    #pragma unroll
    for (int r = 0; r < 8; r++) {
      int row = mB + r;
      if (sValid[w][row]) {
        per_atom[sAtom[w][row]] = sMask[w][row] * (vs[r] + b3v + corr);
      }
    }
  }
}

// ---------- per-molecule reduction: one wave per molecule, fixed order ----------
__global__ void k_red(const float* __restrict__ per_atom, float* __restrict__ out) {
  int w = threadIdx.x >> 5, lane = threadIdx.x & 31;
  int m = blockIdx.x * 8 + w;
  if (m >= NB) return;
  const float* p = per_atom + (long)m * NA;
  float s = p[lane] + p[lane + 32];
  s += __shfl_xor(s, 16, 32);
  s += __shfl_xor(s, 8, 32);
  s += __shfl_xor(s, 4, 32);
  s += __shfl_xor(s, 2, 32);
  s += __shfl_xor(s, 1, 32);
  if (lane == 0) out[m] = s;
}

extern "C" void kernel_launch(void* const* d_in, const int* in_sizes, int n_in,
                              void* d_out, int out_size, void* d_ws, size_t ws_size,
                              hipStream_t stream) {
  const float* x    = (const float*)d_in[0];
  const float* mask = (const float*)d_in[1];
  const float* W1   = (const float*)d_in[2];
  const float* b1   = (const float*)d_in[3];
  const float* W2   = (const float*)d_in[4];
  const float* b2   = (const float*)d_in[5];
  const float* W3   = (const float*)d_in[6];
  const float* b3   = (const float*)d_in[7];
  float* out = (float*)d_out;

  char* ws = (char*)d_ws;
  int*    counts   = (int*)ws;
  int*    bases    = counts + 8;
  int*    cursors  = bases + 8;
  float*  C        = (float*)(cursors + 8);
  float*  S        = C + 8;
  __bf16* W1T      = (__bf16*)(ws + 256);
  __bf16* W2T      = (__bf16*)(ws + 256 + (size_t)NT*NH1*NFP*2);
  int*    list     = (int*)(ws + 256 + (size_t)NT*NH1*NFP*2 + (size_t)NT*NH2*NH1*2);
  float*  per_atom = (float*)((char*)list + (size_t)NATOM*4);

  k_init<<<1, 64, 0, stream>>>(b1, W2, b2, W3, b3, counts, C, S);
  k_w1t<<<(NT*NH1*NFP + 255)/256, 256, 0, stream>>>(W1, W1T);
  k_w2t<<<(NT*NH2*NH1 + 255)/256, 256, 0, stream>>>(W2, W2T);
  k_hist<<<(NATOM + 255)/256, 256, 0, stream>>>(x, counts);
  k_bases<<<1, 1, 0, stream>>>(counts, bases, cursors);
  k_scatter<<<(NATOM + 255)/256, 256, 0, stream>>>(x, cursors, list);
  k_mlp<<<(MAXTILES + NW - 1)/NW, 32*NW, 0, stream>>>(
      x, mask, b1, b2, W3, b3, counts, bases, C, S, W1T, W2T, list, per_atom);
  k_red<<<(NB + 7)/8, 256, 0, stream>>>(per_atom, out);
}
